// BidirRWKV6LucidTimeMix_69818988364372
// MI455X (gfx1250) — compile-verified, matrix operand order checked by reading, not run on hardware
//
#include <hip/hip_runtime.h>
#include <hip/hip_bf16.h>
#include <math.h>

// ---------------- problem constants ----------------
#define BB   4
#define TT   1024
#define DD   768
#define HH   12
#define KK   64
#define DMIXC 32
#define BT   (BB*TT)                       // 4096 tokens
#define BTD  ((size_t)BT*DD)               // 3,145,728
#define MATSZ ((size_t)TT*TT)              // 1,048,576
#define NBH  (BB*HH)                       // 48 batched heads

typedef __attribute__((ext_vector_type(16))) __bf16 v16bf;
typedef __attribute__((ext_vector_type(8)))  __bf16 v8bf;
typedef __attribute__((ext_vector_type(4)))  __bf16 v4bf;
typedef __attribute__((ext_vector_type(8)))  float  v8f;
typedef __attribute__((ext_vector_type(4)))  float  v4f;   // native vec for nontemporal

// =====================================================================
// Kernel 1: token shift, 5-way low-rank mixing, decay MLP.
// =====================================================================
__global__ __launch_bounds__(256)
void mix_kernel(const float* __restrict__ x,
                const float* __restrict__ tmx, const float* __restrict__ tmw,
                const float* __restrict__ tmk, const float* __restrict__ tmv,
                const float* __restrict__ tmr, const float* __restrict__ tmg,
                const float* __restrict__ maa_w1,   // (768,160)
                const float* __restrict__ maa_w2,   // (5,32,768)
                const float* __restrict__ time_decay,
                const float* __restrict__ dw1,      // (768,64)
                const float* __restrict__ dw2,      // (64,768)
                float* __restrict__ XW, float* __restrict__ XK,
                float* __restrict__ XV, float* __restrict__ XR,
                float* __restrict__ XG, float* __restrict__ WNEG)
{
    __shared__ float xs[DD], dxs[DD], xxs[DD];   // xxs reused as xw row later
    __shared__ float ms[5*DMIXC];
    __shared__ float hws[64];
    const int bt = blockIdx.x;
    const int t  = bt % TT;
    const int tid = threadIdx.x;

    for (int d = tid; d < DD; d += 256) {
        float xv = x[(size_t)bt*DD + d];
        float lf = (t > 0)      ? x[(size_t)(bt-1)*DD + d] : 0.0f;
        float rt = (t < TT-1)   ? x[(size_t)(bt+1)*DD + d] : 0.0f;
        float dx = 0.5f*(lf + rt) - xv;
        xs[d] = xv; dxs[d] = dx;
        xxs[d] = xv + dx * tmx[d];
    }
    __syncthreads();
    if (tid < 5*DMIXC) {
        float acc = 0.0f;
        for (int d = 0; d < DD; ++d) acc += xxs[d] * maa_w1[d*(5*DMIXC) + tid];
        ms[tid] = tanhf(acc);
    }
    __syncthreads();
    for (int d = tid; d < DD; d += 256) {
        float mm[5];
        #pragma unroll
        for (int c = 0; c < 5; ++c) {
            float acc = 0.0f;
            for (int e = 0; e < DMIXC; ++e)
                acc += ms[c*DMIXC + e] * maa_w2[(size_t)(c*DMIXC + e)*DD + d];
            mm[c] = acc;
        }
        float xv = xs[d], dx = dxs[d];
        float xw = xv + dx * (tmw[d] + mm[0]);
        XW[(size_t)bt*DD + d] = xw;
        XK[(size_t)bt*DD + d] = xv + dx * (tmk[d] + mm[1]);
        XV[(size_t)bt*DD + d] = xv + dx * (tmv[d] + mm[2]);
        XR[(size_t)bt*DD + d] = xv + dx * (tmr[d] + mm[3]);
        XG[(size_t)bt*DD + d] = xv + dx * (tmg[d] + mm[4]);
        xxs[d] = xw;                                  // reuse as xw row
    }
    __syncthreads();
    if (tid < 64) {
        float acc = 0.0f;
        for (int d = 0; d < DD; ++d) acc += xxs[d] * dw1[d*64 + tid];
        hws[tid] = tanhf(acc);
    }
    __syncthreads();
    for (int d = tid; d < DD; d += 256) {
        float acc = time_decay[d];
        for (int e = 0; e < 64; ++e) acc += hws[e] * dw2[e*DD + d];
        WNEG[(size_t)bt*DD + d] = -expf(acc);         // w_h = -exp(w)
    }
}

// =====================================================================
// Kernel 2: cumsum over T for every (b,d) lane.
// =====================================================================
__global__ void cumsum_kernel(const float* __restrict__ WNEG, float* __restrict__ WHCS)
{
    int idx = blockIdx.x*256 + threadIdx.x;
    if (idx >= BB*DD) return;
    int b = idx / DD, d = idx % DD;
    float cs = 0.0f;
    for (int t = 0; t < TT; ++t) {
        size_t o = ((size_t)(b*TT + t))*DD + d;
        cs += WNEG[o];
        WHCS[o] = cs;
    }
}

// =====================================================================
// Generic batched WMMA GEMM, software-pipelined:
//   iteration i: regs->LDS (drains loadcnt), barrier, issue global
//   fetch of tile i+1 into regs, WMMA-compute tile i from LDS, barrier.
// Block tile 128x64, 8 waves (2Mx4N), 4 accumulators/wave.
// B stored n-major in LDS so both A and B fragments are 2x ds_load_b128.
//   CDNA5 16-bit lane layout: element e -> K = ((e>>3)<<4)+half*8+(e&7)
// ntC=1 streams C with nontemporal stores (T x T outputs bypass-rinse L2).
// =====================================================================
#define GTM 128
#define GTN 64
#define GTK 32
#define LDK 40     // padded row length (bf16); 80B rows keep 16B alignment

__device__ inline v16bf load_frag(const __bf16* base)
{
    v8bf lo = *(const v8bf*)(base);
    v8bf hi = *(const v8bf*)(base + 16);
    return __builtin_shufflevector(lo, hi, 0,1,2,3,4,5,6,7,8,9,10,11,12,13,14,15);
}

__global__ __launch_bounds__(256)
void gemm_bf16(const float* __restrict__ Ag, const float* __restrict__ Bg,
               float* __restrict__ Cg,
               int M, int N, int Kd, int lda, int ldb, int ldc,
               long long sAb, long long sAh, long long sBb, long long sBh,
               long long sCb, long long sCh, int transB, int ntC)
{
    __shared__ __bf16 As[GTM*LDK];   // 10 KB
    __shared__ __bf16 Bt[GTN*LDK];   // n-major, 5 KB
    const int z  = blockIdx.z, zb = z / HH, zh = z % HH;
    const float* A  = Ag + zb*sAb + zh*sAh;
    const float* Bm = Bg + zb*sBb + zh*sBh;
    float*       C  = Cg + zb*sCb + zh*sCh;

    const int tid  = threadIdx.x;
    const int lane = tid & 31, wave = tid >> 5;
    const int wn = wave & 3, wm = wave >> 2;
    const int mBase = blockIdx.y * GTM, nBase = blockIdx.x * GTN;
    const int col  = lane & 15, half = lane >> 4;

    float4 ar[4], br[2];
    auto fetch = [&](int kk) {
        #pragma unroll
        for (int i = 0; i < 4; ++i) {
            int f = tid + i*256;
            int r = f >> 3, c4 = (f & 7) * 4;
            ar[i] = *(const float4*)&A[(size_t)(mBase + r)*lda + (kk + c4)];
        }
        if (transB) {
            #pragma unroll
            for (int i = 0; i < 2; ++i) {
                int f = tid + i*256;
                int n = f >> 3, c4 = (f & 7) * 4;
                br[i] = *(const float4*)&Bm[(size_t)(nBase + n)*ldb + (kk + c4)];
            }
        } else {
            #pragma unroll
            for (int i = 0; i < 2; ++i) {
                int f = tid + i*256;
                int kr = f >> 4, n4 = (f & 15) * 4;
                br[i] = *(const float4*)&Bm[(size_t)(kk + kr)*ldb + (nBase + n4)];
            }
        }
    };
    auto stage = [&]() {
        #pragma unroll
        for (int i = 0; i < 4; ++i) {
            int f = tid + i*256;
            int r = f >> 3, c4 = (f & 7) * 4;
            float4 v = ar[i];
            v4bf hv = { (__bf16)v.x, (__bf16)v.y, (__bf16)v.z, (__bf16)v.w };
            *(v4bf*)&As[r*LDK + c4] = hv;
        }
        if (transB) {
            #pragma unroll
            for (int i = 0; i < 2; ++i) {
                int f = tid + i*256;
                int n = f >> 3, c4 = (f & 7) * 4;
                float4 v = br[i];
                v4bf hv = { (__bf16)v.x, (__bf16)v.y, (__bf16)v.z, (__bf16)v.w };
                *(v4bf*)&Bt[n*LDK + c4] = hv;
            }
        } else {
            #pragma unroll
            for (int i = 0; i < 2; ++i) {
                int f = tid + i*256;
                int kr = f >> 4, n4 = (f & 15) * 4;
                float4 v = br[i];
                Bt[(n4+0)*LDK + kr] = (__bf16)v.x;
                Bt[(n4+1)*LDK + kr] = (__bf16)v.y;
                Bt[(n4+2)*LDK + kr] = (__bf16)v.z;
                Bt[(n4+3)*LDK + kr] = (__bf16)v.w;
            }
        }
    };

    v8f acc[4] = {v8f{}, v8f{}, v8f{}, v8f{}};
    fetch(0);
    for (int k0 = 0; k0 < Kd; k0 += GTK) {
        stage();                      // drains loadcnt for tile k0
        __syncthreads();
        if (k0 + GTK < Kd) fetch(k0 + GTK);   // loads in flight during WMMA
        v16bf bfr = load_frag(&Bt[(wn*16 + col)*LDK + half*8]);
        const __bf16* arow = &As[(wm*64 + col)*LDK + half*8];
        #pragma unroll
        for (int s = 0; s < 4; ++s) {
            v16bf a = load_frag(arow + (s*16)*LDK);
            acc[s] = __builtin_amdgcn_wmma_f32_16x16x32_bf16(
                         false, a, false, bfr, (short)0, acc[s], false, false);
        }
        __syncthreads();
    }

    const int rb = half*8;
    const int cc = nBase + wn*16 + col;
    #pragma unroll
    for (int s = 0; s < 4; ++s) {
        int row = mBase + wm*64 + s*16 + rb;
        #pragma unroll
        for (int v = 0; v < 8; ++v) {
            float* cp = &C[(size_t)(row + v)*ldc + cc];
            if (ntC) __builtin_nontemporal_store(acc[s][v], cp);
            else     *cp = acc[s][v];
        }
    }
}

// =====================================================================
// Kernel: head-space decay scaling + kn normalization.
// =====================================================================
__global__ __launch_bounds__(256)
void head_scale(const float* __restrict__ R, const float* __restrict__ Kp,
                const float* __restrict__ WNEG, const float* __restrict__ WHCS,
                float* __restrict__ RF, float* __restrict__ KF,
                float* __restrict__ RBm, float* __restrict__ KBm,
                float* __restrict__ KN)
{
    __shared__ float nrm[HH];
    const int bt = blockIdx.x;
    const int b  = bt / TT;
    const int tid = threadIdx.x;
    if (tid < HH) nrm[tid] = 0.0f;
    __syncthreads();
    for (int d = tid; d < DD; d += 256) {
        float kv = Kp[(size_t)bt*DD + d];
        atomicAdd(&nrm[d >> 6], kv*kv);
    }
    __syncthreads();
    const size_t midO = ((size_t)(b*TT + TT/2))*DD;
    for (int d = tid; d < DD; d += 256) {
        size_t o = (size_t)bt*DD + d;
        float rv = R[o], kv = Kp[o];
        float cs  = WHCS[o];
        float csb = cs - WNEG[o];
        float csm  = WHCS[midO + d];
        float csbm = csm - WNEG[midO + d];
        float f  = fminf(fmaxf(cs  - csm,  -60.f), 60.f);
        float bb = fminf(fmaxf(csb - csbm, -60.f), 60.f);
        RF[o]  = rv * expf(f);
        KF[o]  = kv * expf(-f);
        RBm[o] = rv * expf(-bb);
        KBm[o] = kv * expf(bb);
        KN[o]  = kv / fmaxf(sqrtf(nrm[d >> 6]), 1e-12f);
    }
}

// =====================================================================
// Kernel: P = I + exp(clip(softplus(temp_h) * gram, -20, 20))
// v4f + nontemporal (streams 201 MB past the L2 working set).
// =====================================================================
__global__ void build_P(const float* __restrict__ gram,
                        const float* __restrict__ lucid,
                        float* __restrict__ Pg)
{
    const size_t nv = ((size_t)NBH * MATSZ) >> 2;
    for (size_t i = (size_t)blockIdx.x*blockDim.x + threadIdx.x;
         i < nv; i += (size_t)gridDim.x*blockDim.x) {
        size_t base = i << 2;
        size_t z = base / MATSZ;
        size_t r = base - z*MATSZ;
        int t = (int)(r >> 10), s0 = (int)(r & 1023);
        float temp = log1pf(expf(lucid[(int)(z % HH)]));
        v4f g4 = __builtin_nontemporal_load((const v4f*)gram + i);
        v4f o4;
        #pragma unroll
        for (int j = 0; j < 4; ++j) {
            float g = fminf(fmaxf(temp * g4[j], -20.f), 20.f);
            o4[j] = expf(g) + ((t == s0 + j) ? 1.0f : 0.0f);
        }
        __builtin_nontemporal_store(o4, (v4f*)Pg + i);
    }
}

// =====================================================================
// Kernel: per-(b,h) dense solve  P * Pv = v_h.  Blocked LU (NB=64):
//   - fp32 panel factorization (no pivoting; diag ~ 1+e^softplus(1))
//   - fp32 row-block triangular solve for U
//   - trailing update via bf16x3 split WMMA:
//       a = a_hi + a_lo, b = b_hi + b_lo,  C -= ah*bh + ah*bl + al*bh
//     (~fp32 accuracy at bf16 WMMA rate; ~NB x less O(T^3) L2 traffic
//      than the rank-1 form). One 8-wave workgroup per system.
// =====================================================================
#define LNB 64
#define LDT 72     // bf16 tile row pitch (144 B, 16B-aligned)

__global__ __launch_bounds__(256)
void lu_solve(float* __restrict__ Pg, const float* __restrict__ Vg,
              float* __restrict__ PVg)
{
    const int z = blockIdx.x, b = z / HH, h = z % HH;
    float* Mt = Pg  + (size_t)z * MATSZ;
    float* PV = PVg + (size_t)z * TT * KK;
    const int tid = threadIdx.x;
    const int lane = tid & 31, wave = tid >> 5;
    const int wn = wave & 3, wm = wave >> 2;
    const int col = lane & 15, half = lane >> 4;

    __shared__ float  prowp[LNB];
    __shared__ float  Lb[LNB*(LNB+1)];               // 16.6 KB
    __shared__ __bf16 Lhi[LNB*LDT], Llo[LNB*LDT];    // 9 KB each
    __shared__ __bf16 Uthi[LNB*LDT], Utlo[LNB*LDT];  // n-major
    __shared__ float  rrow[KK];

    for (int p = 0; p < TT/LNB; ++p) {
        const int j0 = p*LNB;
        // ---- panel factorization (columns j0..j0+63, fp32) ----
        for (int j = j0; j < j0 + LNB; ++j) {
            const int w = j0 + LNB - j;
            if (tid < w) prowp[tid] = Mt[(size_t)j*TT + j + tid];
            __syncthreads();
            const float pinv = 1.0f / prowp[0];
            for (int i = j + 1 + tid; i < TT; i += 256) {
                float* Mi = Mt + (size_t)i*TT;
                float lij = Mi[j] * pinv;
                Mi[j] = lij;
                for (int c = 1; c < w; ++c) Mi[j + c] -= lij * prowp[c];
            }
            __syncthreads();
        }

        const int rem = TT - j0 - LNB;
        if (rem <= 0) continue;

        // ---- stage unit-lower diag block into LDS ----
        for (int e = tid; e < LNB*LNB; e += 256) {
            int r = e >> 6, c = e & 63;
            Lb[r*(LNB+1) + c] = Mt[(size_t)(j0 + r)*TT + j0 + c];
        }
        __syncthreads();

        // ---- U row-block solve: rows j0..j0+63, cols j0+64.. ----
        for (int jj = 1; jj < LNB; ++jj) {
            for (int c = j0 + LNB + tid; c < TT; c += 256) {
                float acc = Mt[(size_t)(j0 + jj)*TT + c];
                for (int i = 0; i < jj; ++i)
                    acc -= Lb[jj*(LNB+1) + i] * Mt[(size_t)(j0 + i)*TT + c];
                Mt[(size_t)(j0 + jj)*TT + c] = acc;
            }
            __syncthreads();
        }

        // ---- trailing update with bf16x3 split WMMA ----
        const int ntl = rem / LNB;
        for (int ti = 0; ti < ntl; ++ti) {
            const int i0 = j0 + LNB + ti*LNB;
            __syncthreads();
            // stage L tile (rows i0.., cols j0..) hi/lo
            #pragma unroll
            for (int i = 0; i < 4; ++i) {
                int f = tid + i*256;
                int r = f >> 4, c4 = (f & 15) * 4;
                float4 v = *(const float4*)&Mt[(size_t)(i0 + r)*TT + j0 + c4];
                v4bf hv = { (__bf16)v.x, (__bf16)v.y, (__bf16)v.z, (__bf16)v.w };
                v4bf lv = { (__bf16)(v.x - (float)hv[0]), (__bf16)(v.y - (float)hv[1]),
                            (__bf16)(v.z - (float)hv[2]), (__bf16)(v.w - (float)hv[3]) };
                *(v4bf*)&Lhi[r*LDT + c4] = hv;
                *(v4bf*)&Llo[r*LDT + c4] = lv;
            }
            for (int tj = 0; tj < ntl; ++tj) {
                const int c0 = j0 + LNB + tj*LNB;
                __syncthreads();
                // stage U tile (rows j0.., cols c0..) hi/lo, n-major
                #pragma unroll
                for (int i = 0; i < 4; ++i) {
                    int f = tid + i*256;
                    int q = f >> 4, n4 = (f & 15) * 4;
                    float4 v = *(const float4*)&Mt[(size_t)(j0 + q)*TT + c0 + n4];
                    float vv[4] = { v.x, v.y, v.z, v.w };
                    #pragma unroll
                    for (int jx = 0; jx < 4; ++jx) {
                        __bf16 hv = (__bf16)vv[jx];
                        Uthi[(n4 + jx)*LDT + q] = hv;
                        Utlo[(n4 + jx)*LDT + q] = (__bf16)(vv[jx] - (float)hv);
                    }
                }
                __syncthreads();
                // compute 64x64 tile: 8 waves, 2 subtiles each, K=64 (2 steps)
                v8f a0 = {}, a1 = {};
                #pragma unroll
                for (int ks = 0; ks < 2; ++ks) {
                    const int kb = ks*32 + half*8;
                    v16bf bh = load_frag(&Uthi[(wn*16 + col)*LDT + kb]);
                    v16bf bl = load_frag(&Utlo[(wn*16 + col)*LDT + kb]);
                    #pragma unroll
                    for (int s = 0; s < 2; ++s) {
                        const int rl = (wm*32 + s*16 + col)*LDT + kb;
                        v16bf ah = load_frag(&Lhi[rl]);
                        v16bf al = load_frag(&Llo[rl]);
                        v8f* ac = s ? &a1 : &a0;
                        *ac = __builtin_amdgcn_wmma_f32_16x16x32_bf16(
                                  false, ah, false, bh, (short)0, *ac, false, false);
                        *ac = __builtin_amdgcn_wmma_f32_16x16x32_bf16(
                                  false, ah, false, bl, (short)0, *ac, false, false);
                        *ac = __builtin_amdgcn_wmma_f32_16x16x32_bf16(
                                  false, al, false, bh, (short)0, *ac, false, false);
                    }
                }
                // C -= acc
                const int rb = half*8, cc = c0 + wn*16 + col;
                #pragma unroll
                for (int s = 0; s < 2; ++s) {
                    v8f ac = s ? a1 : a0;
                    int row = i0 + wm*32 + s*16 + rb;
                    #pragma unroll
                    for (int v = 0; v < 8; ++v)
                        Mt[(size_t)(row + v)*TT + cc] -= ac[v];
                }
            }
        }
        __syncthreads();
    }

    // ---- RHS: load head view of V ----
    for (int e = tid; e < TT*KK; e += 256) {
        int t = e >> 6, c = e & 63;
        PV[e] = Vg[((size_t)(b*TT + t))*DD + h*KK + c];
    }
    __syncthreads();

    const int ci = tid & 63, ri = tid >> 6;          // 4 row groups x 64 cols
    // forward substitution (unit lower)
    for (int j = 0; j < TT-1; ++j) {
        if (tid < KK) rrow[tid] = PV[j*KK + tid];
        __syncthreads();
        for (int i = j + 1 + ri; i < TT; i += 4)
            PV[i*KK + ci] -= Mt[(size_t)i*TT + j] * rrow[ci];
        __syncthreads();
    }
    // back substitution
    for (int j = TT-1; j >= 0; --j) {
        if (tid < KK) {
            float v = PV[j*KK + tid] / Mt[(size_t)j*TT + j];
            rrow[tid] = v;
            PV[j*KK + tid] = v;
        }
        __syncthreads();
        for (int i = ri; i < j; i += 4)
            PV[i*KK + ci] -= Mt[(size_t)i*TT + j] * rrow[ci];
        __syncthreads();
    }
}

// =====================================================================
// Kernel: A = tril(fwd) + triu(bwd,1); v4f + nontemporal streaming.
// =====================================================================
__global__ void combine_A(const float* __restrict__ fwd, float* __restrict__ Ag)
{
    const size_t nv = ((size_t)NBH * MATSZ) >> 2;
    for (size_t i = (size_t)blockIdx.x*blockDim.x + threadIdx.x;
         i < nv; i += (size_t)gridDim.x*blockDim.x) {
        size_t base = i << 2;
        size_t r = base % MATSZ;
        int t = (int)(r >> 10), s0 = (int)(r & 1023);
        v4f f4 = __builtin_nontemporal_load((const v4f*)fwd + i);
        v4f a4 = __builtin_nontemporal_load((const v4f*)Ag + i);
        #pragma unroll
        for (int j = 0; j < 4; ++j)
            if (s0 + j <= t) a4[j] = f4[j];
        __builtin_nontemporal_store(a4, (v4f*)Ag + i);
    }
}

// =====================================================================
// Kernel: per-head GroupNorm (eps = 1e-5*64) + SiLU gate.
// =====================================================================
__global__ __launch_bounds__(256)
void groupnorm_gate(const float* __restrict__ Y, const float* __restrict__ G,
                    const float* __restrict__ lnw, const float* __restrict__ lnb,
                    float* __restrict__ YG)
{
    __shared__ float ssum[HH], ssq[HH];
    const int bt = blockIdx.x, b = bt / TT, t = bt % TT;
    const int tid = threadIdx.x;
    if (tid < HH) { ssum[tid] = 0.0f; ssq[tid] = 0.0f; }
    __syncthreads();
    for (int d = tid; d < DD; d += 256) {
        int h = d >> 6, kk = d & 63;
        float y = Y[(((size_t)(b*HH + h))*TT + t)*KK + kk];
        atomicAdd(&ssum[h], y);
        atomicAdd(&ssq[h],  y*y);
    }
    __syncthreads();
    const float eps = 1e-5f * 64.0f;
    for (int d = tid; d < DD; d += 256) {
        int h = d >> 6, kk = d & 63;
        float y  = Y[(((size_t)(b*HH + h))*TT + t)*KK + kk];
        float mu = ssum[h] * (1.0f/64.0f);
        float var = ssq[h] * (1.0f/64.0f) - mu*mu;
        float yn = (y - mu) * rsqrtf(var + eps);
        yn = yn * lnw[d] + lnb[d];
        float gv = G[(size_t)bt*DD + d];
        float gate = gv / (1.0f + expf(-gv));        // silu
        YG[(size_t)bt*DD + d] = yn * gate;
    }
}

// =====================================================================
// host launcher
// =====================================================================
extern "C" void kernel_launch(void* const* d_in, const int* in_sizes, int n_in,
                              void* d_out, int out_size, void* d_ws, size_t ws_size,
                              hipStream_t stream)
{
    const float* x    = (const float*)d_in[0];
    const float* tmx  = (const float*)d_in[1];
    const float* tmw  = (const float*)d_in[2];
    const float* tmk  = (const float*)d_in[3];
    const float* tmv  = (const float*)d_in[4];
    const float* tmr  = (const float*)d_in[5];
    const float* tmg  = (const float*)d_in[6];
    const float* maa_w1 = (const float*)d_in[7];
    const float* maa_w2 = (const float*)d_in[8];
    const float* tdec = (const float*)d_in[9];
    const float* dw1  = (const float*)d_in[10];
    const float* dw2  = (const float*)d_in[11];
    const float* lucid= (const float*)d_in[12];
    const float* Wr   = (const float*)d_in[13];
    const float* Wk   = (const float*)d_in[14];
    const float* Wv   = (const float*)d_in[15];
    const float* Wg   = (const float*)d_in[16];
    const float* Wo   = (const float*)d_in[17];
    const float* lnw  = (const float*)d_in[18];
    const float* lnb  = (const float*)d_in[19];
    float* out = (float*)d_out;

    char* wp = (char*)d_ws;
    auto alloc = [&](size_t nfloats) {
        float* p = (float*)wp;
        wp += ((nfloats*sizeof(float) + 255) / 256) * 256;
        return p;
    };
    float* XW = alloc(BTD); float* XK = alloc(BTD); float* XV = alloc(BTD);
    float* XR = alloc(BTD); float* XG = alloc(BTD);
    float* R  = alloc(BTD); float* Kp = alloc(BTD); float* V  = alloc(BTD);
    float* G  = alloc(BTD);
    float* WNEG = alloc(BTD); float* WHCS = alloc(BTD);
    float* RF = alloc(BTD); float* KF = alloc(BTD);
    float* RBm= alloc(BTD); float* KBm= alloc(BTD);
    float* KN = alloc(BTD);
    float* PV = alloc(BTD); float* Y  = alloc(BTD); float* YG = alloc(BTD);
    float* TMP = alloc((size_t)NBH * MATSZ);
    float* Abuf= alloc((size_t)NBH * MATSZ);
    float* Pm  = alloc((size_t)NBH * MATSZ);

    auto gemm = [&](const float* A, const float* Bm, float* C,
                    int M, int N, int Kd, int lda, int ldb, int ldc,
                    long long sAb, long long sAh, long long sBb, long long sBh,
                    long long sCb, long long sCh, int transB, int ntC, int batch) {
        dim3 grid(N/GTN, M/GTM, batch);
        gemm_bf16<<<grid, 256, 0, stream>>>(A, Bm, C, M, N, Kd, lda, ldb, ldc,
                                            sAb, sAh, sBb, sBh, sCb, sCh, transB, ntC);
    };

    // 1) mixing + decay
    mix_kernel<<<BT, 256, 0, stream>>>(x, tmx, tmw, tmk, tmv, tmr, tmg,
                                       maa_w1, maa_w2, tdec, dw1, dw2,
                                       XW, XK, XV, XR, XG, WNEG);
    // 2) cumsum over T
    cumsum_kernel<<<(BB*DD + 255)/256, 256, 0, stream>>>(WNEG, WHCS);
    // 3) projections (WMMA)
    gemm(XR, Wr, R,  BT, DD, DD, DD, DD, DD, 0,0,0,0,0,0, 0, 0, 1);
    gemm(XK, Wk, Kp, BT, DD, DD, DD, DD, DD, 0,0,0,0,0,0, 0, 0, 1);
    gemm(XV, Wv, V,  BT, DD, DD, DD, DD, DD, 0,0,0,0,0,0, 0, 0, 1);
    gemm(XG, Wg, G,  BT, DD, DD, DD, DD, DD, 0,0,0,0,0,0, 0, 0, 1);
    // 4) decay scaling + kn
    head_scale<<<BT, 256, 0, stream>>>(R, Kp, WNEG, WHCS, RF, KF, RBm, KBm, KN);
    // 5) gram = kn . kn^T (batched, transB, streamed C)
    gemm(KN, KN, TMP, TT, TT, KK, DD, DD, TT,
         (long long)TT*DD, KK, (long long)TT*DD, KK,
         (long long)HH*MATSZ, (long long)MATSZ, 1, 1, NBH);
    // 6) P = I + exp(clip(temp*gram))
    build_P<<<4096, 256, 0, stream>>>(TMP, lucid, Pm);
    // 7) dense solve per head (blocked LU + split-WMMA trailing update)
    lu_solve<<<NBH, 256, 0, stream>>>(Pm, V, PV);
    // 8) A_fwd -> TMP ; A_bwd -> Abuf (batched, transB, streamed C)
    gemm(RF,  KF,  TMP,  TT, TT, KK, DD, DD, TT,
         (long long)TT*DD, KK, (long long)TT*DD, KK,
         (long long)HH*MATSZ, (long long)MATSZ, 1, 1, NBH);
    gemm(RBm, KBm, Abuf, TT, TT, KK, DD, DD, TT,
         (long long)TT*DD, KK, (long long)TT*DD, KK,
         (long long)HH*MATSZ, (long long)MATSZ, 1, 1, NBH);
    // 9) A = tril(fwd) + triu(bwd,1)
    combine_A<<<4096, 256, 0, stream>>>(TMP, Abuf);
    // 10) y = A @ Pv (batched)
    gemm(Abuf, PV, Y, TT, KK, TT, TT, KK, KK,
         (long long)HH*MATSZ, (long long)MATSZ,
         (long long)HH*TT*KK, (long long)TT*KK,
         (long long)HH*TT*KK, (long long)TT*KK, 0, 0, NBH);
    // 11) GroupNorm + SiLU gate
    groupnorm_gate<<<BT, 256, 0, stream>>>(Y, G, lnw, lnb, YG);
    // 12) output projection
    gemm(YG, Wo, out, BT, DD, DD, DD, DD, DD, 0,0,0,0,0,0, 0, 0, 1);
}
